// Processor_50775103373539
// MI455X (gfx1250) — matrix operand order repaired and checked by reading, not compiled, not run on hardware
//
#include <hip/hip_runtime.h>

#define LATENT 128
#define STEPS 5
#define N_MESH_C 100000
#define N_OBJ_C 1024
#define E_MO_C 400000
#define E_OM_C 400000
#define LN_EPS 1e-5f

typedef __attribute__((ext_vector_type(16))) _Float16 v16h;
typedef __attribute__((ext_vector_type(8)))  _Float16 v8h;
typedef __attribute__((ext_vector_type(8)))  float    v8f;

static constexpr int ROWS = 64;                 // edges/nodes per block
static constexpr int SH   = LATENT + 8;         // 136 (hidden LDS stride, halves/floats)
static constexpr int SX_E = 3 * LATENT + 8;     // 392
static constexpr int SX_N = 2 * LATENT + 8;     // 264

static constexpr size_t LDS_EDGE = (size_t)ROWS * SX_E * 2 + 2 * (size_t)ROWS * SH * 2 + (size_t)ROWS * SH * 4;
static constexpr size_t LDS_NODE = (size_t)ROWS * SX_N * 2 + 2 * (size_t)ROWS * SH * 2 + (size_t)ROWS * SH * 4;

union V16H { v16h v; v8h h[2]; };

// ---- WMMA fragment loaders ---------------------------------------------------
// A (16x32 f16): lane L<16 -> row L, K = {kc..kc+7, kc+16..kc+23}
//                lane L>=16 -> row L-16, K = {kc+8..kc+15, kc+24..kc+31}
__device__ __forceinline__ v16h ldsA(const _Float16* base /*[row0][kc]*/, int sx, int lane) {
    int r = lane & 15;
    int koff = (lane >> 4) << 3;  // 0 or 8
    const _Float16* p = base + r * sx + koff;
    V16H u;
    u.h[0] = *(const v8h*)(p);
    u.h[1] = *(const v8h*)(p + 16);
    return u.v;
}

// B (32x16 f16): lane L<16 -> col L, K = {kc..kc+7, kc+16..kc+23} ; lane L>=16 -> col L-16, K = {kc+8..kc+15, kc+24..kc+31}
// Weights stored transposed f16: Wt[n][k], stride K, so each lane loads two runs of 8 contiguous halves.
__device__ __forceinline__ v16h gloB(const _Float16* __restrict__ wt, int K, int nbase, int kc, int lane) {
    int n = nbase + (lane & 15);
    int koff = (lane >> 4) << 3;  // 0 or 8
    const _Float16* p = wt + (size_t)n * K + kc + koff;
    V16H u;
    u.h[0] = *(const v8h*)(p);
    u.h[1] = *(const v8h*)(p + 16);
    return u.v;
}

// One wave computes rows [row0, row0+16) x 128 columns, accumulating over K.
// Software-pipelined: B fragments for chunk k+1 are in flight while chunk k's
// 8 WMMAs issue back-to-back (shared A, distinct B/acc -> no WMMA-WMMA hazard).
template<int K>
__device__ __forceinline__ void wave_gemm(const _Float16* Xh, int sx, int row0,
                                          const _Float16* __restrict__ Wt,
                                          v8f acc[8], int lane) {
#pragma unroll
    for (int t = 0; t < 8; ++t) {
#pragma unroll
        for (int i = 0; i < 8; ++i) acc[t][i] = 0.0f;
    }

    const _Float16* arow = Xh + (size_t)row0 * sx;
    v16h a = ldsA(arow, sx, lane);
    v16h b[8];
#pragma unroll
    for (int t = 0; t < 8; ++t) b[t] = gloB(Wt, K, t * 16, 0, lane);

#pragma unroll
    for (int kc = 0; kc < K; kc += 32) {
        const int kn = (kc + 32 < K) ? (kc + 32) : 0;  // wrapped prefetch; dead-coded on last iter
        v16h an = ldsA(arow + kn, sx, lane);
        v16h bn[8];
#pragma unroll
        for (int t = 0; t < 8; ++t) bn[t] = gloB(Wt, K, t * 16, kn, lane);
#pragma unroll
        for (int t = 0; t < 8; ++t)
            acc[t] = __builtin_amdgcn_wmma_f32_16x16x32_f16(
                false, a, false, b[t], (short)0, acc[t], false, false);
        a = an;
#pragma unroll
        for (int t = 0; t < 8; ++t) b[t] = bn[t];
    }
}

// C/D layout: lane L<16 -> col L, rows v (VGPR v = row v) ; lane>=16 -> rows v+8.
__device__ __forceinline__ void store_relu_f16(const v8f acc[8], _Float16* H, int row0,
                                               const float* __restrict__ bias, int lane) {
    int nl = lane & 15;
    int rb = row0 + ((lane >> 4) << 3);
#pragma unroll
    for (int t = 0; t < 8; ++t) {
        int n = t * 16 + nl;
        float b = bias[n];
#pragma unroll
        for (int v = 0; v < 8; ++v) {
            float x = acc[t][v] + b;
            x = x > 0.0f ? x : 0.0f;
            H[(size_t)(rb + v) * SH + n] = (_Float16)x;
        }
    }
}

__device__ __forceinline__ void store_f32(const v8f acc[8], float* H, int row0,
                                          const float* __restrict__ bias, int lane) {
    int nl = lane & 15;
    int rb = row0 + ((lane >> 4) << 3);
#pragma unroll
    for (int t = 0; t < 8; ++t) {
        int n = t * 16 + nl;
        float b = bias[n];
#pragma unroll
        for (int v = 0; v < 8; ++v)
            H[(size_t)(rb + v) * SH + n] = acc[t][v] + b;
    }
}

// relu(X@W1+b1) -> relu(@W2+b2) -> @W3+b3 (f32 in Hf). All 4 waves participate.
template<int K1>
__device__ __forceinline__ void mlp3(const _Float16* Xh, int sx,
                                     _Float16* Ha, _Float16* Hb, float* Hf,
                                     const _Float16* W1t, const _Float16* W2t, const _Float16* W3t,
                                     const float* b1, const float* b2, const float* b3,
                                     int lane, int wave) {
    int row0 = wave * 16;
    v8f acc[8];
    wave_gemm<K1>(Xh, sx, row0, W1t, acc, lane);
    store_relu_f16(acc, Ha, row0, b1, lane);
    __syncthreads();
    wave_gemm<LATENT>(Ha, SH, row0, W2t, acc, lane);
    store_relu_f16(acc, Hb, row0, b2, lane);
    __syncthreads();
    wave_gemm<LATENT>(Hb, SH, row0, W3t, acc, lane);
    store_f32(acc, Hf, row0, b3, lane);
    __syncthreads();
}

// Pairwise LayerNorm stats: thread pair (2r, 2r+1) owns row r, 64 columns each.
__device__ __forceinline__ void ln_stats(const float* __restrict__ h64,
                                         float& mu, float& inv) {
    float s = 0.0f, ss = 0.0f;
#pragma unroll 8
    for (int n = 0; n < 64; ++n) { float x = h64[n]; s += x; ss += x * x; }
    s  += __shfl_xor(s, 1, 32);
    ss += __shfl_xor(ss, 1, 32);
    mu = s * (1.0f / LATENT);
    float var = ss * (1.0f / LATENT) - mu * mu;
    inv = rsqrtf(var + LN_EPS);
}

// ---- Edge kernel: message MLP + LN + residual + scatter-add -----------------
__global__ __launch_bounds__(128)
void edge_mlp_kernel(const float* __restrict__ x_src, const float* __restrict__ x_dst,
                     const int* __restrict__ ei, int E,
                     float* __restrict__ e_attr,   // workspace copy, updated in place
                     float* __restrict__ aggr,     // pre-zeroed, scatter-add dst
                     const _Float16* __restrict__ W1t, const _Float16* __restrict__ W2t,
                     const _Float16* __restrict__ W3t,
                     const float* __restrict__ b1, const float* __restrict__ b2,
                     const float* __restrict__ b3,
                     const float* __restrict__ ln_g, const float* __restrict__ ln_b) {
    extern __shared__ char smem_raw[];
    _Float16* Xh = (_Float16*)smem_raw;
    _Float16* Ha = Xh + (size_t)ROWS * SX_E;
    _Float16* Hb = Ha + (size_t)ROWS * SH;
    float*    Hf = (float*)(Hb + (size_t)ROWS * SH);

    const int tid = threadIdx.x;
    const int lane = tid & 31, wave = tid >> 5;
    const int eb = blockIdx.x * ROWS;
    const int* srcIdx = ei;        // edge_index[0] = src
    const int* dstIdx = ei + E;    // edge_index[1] = dst

    // Gather cat(x_dst[col], x_src[row], e_attr) -> f16 LDS, 4 floats/iter.
    const int CPR = (3 * LATENT) / 4;  // 96 chunks per row
    for (int idx = tid; idx < ROWS * CPR; idx += 128) {
        int r = idx / CPR;
        int c = (idx - r * CPR) * 4;
        int eg = eb + r;
        const float* src;
        if (c < LATENT)          src = x_dst + (size_t)dstIdx[eg] * LATENT + c;
        else if (c < 2 * LATENT) src = x_src + (size_t)srcIdx[eg] * LATENT + (c - LATENT);
        else                     src = e_attr + (size_t)eg * LATENT + (c - 2 * LATENT);
        float4 v = *(const float4*)src;
        _Float16* d = Xh + (size_t)r * SX_E + c;
        d[0] = (_Float16)v.x; d[1] = (_Float16)v.y; d[2] = (_Float16)v.z; d[3] = (_Float16)v.w;
    }
    __syncthreads();

    mlp3<3 * LATENT>(Xh, SX_E, Ha, Hb, Hf, W1t, W2t, W3t, b1, b2, b3, lane, wave);

    // LayerNorm + residual edge update + scatter-add into aggr[dst].
    // 2 threads per row; each owns 64 columns.
    {
        int r = tid >> 1, half = tid & 1, c0 = half * 64;
        int eg = eb + r;
        const float* h = Hf + (size_t)r * SH + c0;
        float mu, inv;
        ln_stats(h, mu, inv);
        int dn = dstIdx[eg];
        float* arow = aggr + (size_t)dn * LATENT + c0;
        float* erow = e_attr + (size_t)eg * LATENT + c0;
        const float* gg = ln_g + c0;
        const float* bb = ln_b + c0;
        for (int n = 0; n < 64; ++n) {
            float y = (h[n] - mu) * inv * gg[n] + bb[n];
            atomicAdd(arow + n, y);
            erow[n] = erow[n] + y;
        }
    }
}

// ---- Node kernel: node MLP + LN + residual ----------------------------------
__global__ __launch_bounds__(128)
void node_mlp_kernel(const float* __restrict__ x_old, const float* __restrict__ aggr,
                     float* __restrict__ x_new, int Nn,
                     const _Float16* __restrict__ W1t, const _Float16* __restrict__ W2t,
                     const _Float16* __restrict__ W3t,
                     const float* __restrict__ b1, const float* __restrict__ b2,
                     const float* __restrict__ b3,
                     const float* __restrict__ ln_g, const float* __restrict__ ln_b) {
    extern __shared__ char smem_raw[];
    _Float16* Xh = (_Float16*)smem_raw;
    _Float16* Ha = Xh + (size_t)ROWS * SX_N;
    _Float16* Hb = Ha + (size_t)ROWS * SH;
    float*    Hf = (float*)(Hb + (size_t)ROWS * SH);

    const int tid = threadIdx.x;
    const int lane = tid & 31, wave = tid >> 5;
    const int nb = blockIdx.x * ROWS;

    const int CPR = (2 * LATENT) / 4;  // 64 chunks per row
    for (int idx = tid; idx < ROWS * CPR; idx += 128) {
        int r = idx / CPR;
        int c = (idx - r * CPR) * 4;
        int g = nb + r;
        int gs = g < Nn ? g : 0;  // clamp (result discarded for OOB rows)
        const float* src = (c < LATENT) ? x_old + (size_t)gs * LATENT + c
                                        : aggr + (size_t)gs * LATENT + (c - LATENT);
        float4 v = *(const float4*)src;
        _Float16* d = Xh + (size_t)r * SX_N + c;
        d[0] = (_Float16)v.x; d[1] = (_Float16)v.y; d[2] = (_Float16)v.z; d[3] = (_Float16)v.w;
    }
    __syncthreads();

    mlp3<2 * LATENT>(Xh, SX_N, Ha, Hb, Hf, W1t, W2t, W3t, b1, b2, b3, lane, wave);

    {
        int r = tid >> 1, half = tid & 1, c0 = half * 64;
        int g = nb + r;
        if (g < Nn) {
            const float* h = Hf + (size_t)r * SH + c0;
            float mu, inv;
            ln_stats(h, mu, inv);
            const float* xo = x_old + (size_t)g * LATENT + c0;
            float* xn = x_new + (size_t)g * LATENT + c0;
            const float* gg = ln_g + c0;
            const float* bb = ln_b + c0;
            for (int n = 0; n < 64; ++n) {
                float y = (h[n] - mu) * inv * gg[n] + bb[n];
                xn[n] = xo[n] + y;
            }
        }
    }
}

// ---- Weight prep: W[k][128] f32 -> Wt[n][K] f16 -----------------------------
__global__ void wt_prep_kernel(const float* __restrict__ W, _Float16* __restrict__ Wt, int K) {
    int i = blockIdx.x * blockDim.x + threadIdx.x;
    if (i < K * LATENT) {
        int k = i >> 7;            // / 128
        int n = i & (LATENT - 1);  // % 128
        Wt[(size_t)n * K + k] = (_Float16)W[i];
    }
}

// ---- Host orchestration ------------------------------------------------------
static inline size_t align_up(size_t x, size_t a) { return (x + a - 1) / a * a; }

extern "C" void kernel_launch(void* const* d_in, const int* in_sizes, int n_in,
                              void* d_out, int out_size, void* d_ws, size_t ws_size,
                              hipStream_t stream) {
    (void)in_sizes; (void)n_in; (void)out_size; (void)ws_size;

    const float* x_mesh_in = (const float*)d_in[0];
    const float* x_obj_in  = (const float*)d_in[1];
    const int*   ei_mo     = (const int*)d_in[2];
    const int*   ei_om     = (const int*)d_in[3];
    const float* ea_mo     = (const float*)d_in[4];
    const float* ea_om     = (const float*)d_in[5];

    // params leaves (jax tree order, dict keys sorted):
    // net 0 = mo.edge, 1 = mo.node, 2 = om.edge, 3 = om.node
    // per net: W1, W2, W3, b1, b2, b3, ln_b, ln_g  (each with leading STEPS dim)
    const float* leaves[32];
    for (int i = 0; i < 32; ++i) leaves[i] = (const float*)d_in[6 + i];
    const int net_din[4] = {3 * LATENT, 2 * LATENT, 3 * LATENT, 2 * LATENT};

    // Workspace carve
    char* p = (char*)d_ws;
    auto carve = [&](size_t bytes) { char* r = p; p += align_up(bytes, 256); return r; };
    float* e_mo_ws = (float*)carve((size_t)E_MO_C * LATENT * 4);
    float* e_om_ws = (float*)carve((size_t)E_OM_C * LATENT * 4);
    float* xm[2] = {(float*)carve((size_t)N_MESH_C * LATENT * 4),
                    (float*)carve((size_t)N_MESH_C * LATENT * 4)};
    float* xo[2] = {(float*)carve((size_t)N_OBJ_C * LATENT * 4),
                    (float*)carve((size_t)N_OBJ_C * LATENT * 4)};
    float* ag_m = (float*)carve((size_t)N_MESH_C * LATENT * 4);
    float* ag_o = (float*)carve((size_t)N_OBJ_C * LATENT * 4);

    // f16 transposed weights: per step, nets 0..3; per net W1t,W2t,W3t
    const size_t w1h[4] = {(size_t)3 * LATENT * LATENT, (size_t)2 * LATENT * LATENT,
                           (size_t)3 * LATENT * LATENT, (size_t)2 * LATENT * LATENT};
    const size_t whh = (size_t)LATENT * LATENT;  // W2/W3 halves
    size_t per_net[4], per_step_h = 0;
    for (int n2 = 0; n2 < 4; ++n2) { per_net[n2] = w1h[n2] + 2 * whh; per_step_h += per_net[n2]; }
    _Float16* wbase = (_Float16*)carve(STEPS * per_step_h * 2);

    auto wt_ptr = [&](int s, int net, int which) -> _Float16* {
        size_t off = (size_t)s * per_step_h;
        for (int n2 = 0; n2 < net; ++n2) off += per_net[n2];
        if (which >= 1) off += w1h[net];
        if (which >= 2) off += whh;
        return wbase + off;
    };
    auto leaf = [&](int net, int li, int s, size_t per_step_elems) -> const float* {
        return leaves[net * 8 + li] + (size_t)s * per_step_elems;
    };

    // 1) Weight prep (transpose + f32->f16)
    for (int s = 0; s < STEPS; ++s) {
        for (int net = 0; net < 4; ++net) {
            int din = net_din[net];
            int g1 = (din * LATENT + 255) / 256;
            int g2 = (LATENT * LATENT + 255) / 256;
            wt_prep_kernel<<<g1, 256, 0, stream>>>(leaf(net, 0, s, (size_t)din * LATENT), wt_ptr(s, net, 0), din);
            wt_prep_kernel<<<g2, 256, 0, stream>>>(leaf(net, 1, s, whh), wt_ptr(s, net, 1), LATENT);
            wt_prep_kernel<<<g2, 256, 0, stream>>>(leaf(net, 2, s, whh), wt_ptr(s, net, 2), LATENT);
        }
    }

    // 2) Copy mutable state into workspace (inputs must stay pristine)
    hipMemcpyAsync(e_mo_ws, ea_mo, (size_t)E_MO_C * LATENT * 4, hipMemcpyDeviceToDevice, stream);
    hipMemcpyAsync(e_om_ws, ea_om, (size_t)E_OM_C * LATENT * 4, hipMemcpyDeviceToDevice, stream);
    hipMemcpyAsync(xm[0], x_mesh_in, (size_t)N_MESH_C * LATENT * 4, hipMemcpyDeviceToDevice, stream);
    hipMemcpyAsync(xo[0], x_obj_in, (size_t)N_OBJ_C * LATENT * 4, hipMemcpyDeviceToDevice, stream);

    int cur = 0;
    for (int s = 0; s < STEPS; ++s) {
        int nxt = cur ^ 1;
        // --- mo: mesh(src) -> obj(dst) ---
        hipMemsetAsync(ag_o, 0, (size_t)N_OBJ_C * LATENT * 4, stream);
        edge_mlp_kernel<<<E_MO_C / ROWS, 128, LDS_EDGE, stream>>>(
            xm[cur], xo[cur], ei_mo, E_MO_C, e_mo_ws, ag_o,
            wt_ptr(s, 0, 0), wt_ptr(s, 0, 1), wt_ptr(s, 0, 2),
            leaf(0, 3, s, LATENT), leaf(0, 4, s, LATENT), leaf(0, 5, s, LATENT),
            leaf(0, 7, s, LATENT), leaf(0, 6, s, LATENT));  // ln_g, ln_b
        node_mlp_kernel<<<(N_OBJ_C + ROWS - 1) / ROWS, 128, LDS_NODE, stream>>>(
            xo[cur], ag_o, xo[nxt], N_OBJ_C,
            wt_ptr(s, 1, 0), wt_ptr(s, 1, 1), wt_ptr(s, 1, 2),
            leaf(1, 3, s, LATENT), leaf(1, 4, s, LATENT), leaf(1, 5, s, LATENT),
            leaf(1, 7, s, LATENT), leaf(1, 6, s, LATENT));
        // --- om: obj(src) -> mesh(dst), reads pre-step state ---
        hipMemsetAsync(ag_m, 0, (size_t)N_MESH_C * LATENT * 4, stream);
        edge_mlp_kernel<<<E_OM_C / ROWS, 128, LDS_EDGE, stream>>>(
            xo[cur], xm[cur], ei_om, E_OM_C, e_om_ws, ag_m,
            wt_ptr(s, 2, 0), wt_ptr(s, 2, 1), wt_ptr(s, 2, 2),
            leaf(2, 3, s, LATENT), leaf(2, 4, s, LATENT), leaf(2, 5, s, LATENT),
            leaf(2, 7, s, LATENT), leaf(2, 6, s, LATENT));
        node_mlp_kernel<<<(N_MESH_C + ROWS - 1) / ROWS, 128, LDS_NODE, stream>>>(
            xm[cur], ag_m, xm[nxt], N_MESH_C,
            wt_ptr(s, 3, 0), wt_ptr(s, 3, 1), wt_ptr(s, 3, 2),
            leaf(3, 3, s, LATENT), leaf(3, 4, s, LATENT), leaf(3, 5, s, LATENT),
            leaf(3, 7, s, LATENT), leaf(3, 6, s, LATENT));
        cur = nxt;
    }

    // 3) Output: (x_mesh, x_obj) concatenated
    hipMemcpyAsync(d_out, xm[cur], (size_t)N_MESH_C * LATENT * 4, hipMemcpyDeviceToDevice, stream);
    hipMemcpyAsync((float*)d_out + (size_t)N_MESH_C * LATENT, xo[cur],
                   (size_t)N_OBJ_C * LATENT * 4, hipMemcpyDeviceToDevice, stream);
}
